// EPFFNLayer_8469675508185
// MI455X (gfx1250) — matrix operand order repaired and checked by reading, not compiled, not run on hardware
//
#include <hip/hip_runtime.h>

// ---------------------------------------------------------------------------
// MoE expert-parallel FFN for MI455X (gfx1250, wave32, WMMA).
// out[t,:] = sum_e w_e[t] * ( silu(X Wg_e^T) * (X Wu_e^T) ) Wd_e^T
// Compute-bound (~103 GFLOP vs ~112 MB HBM): bf16 16x16x32 WMMA, f32 accum.
// Block tile 128x64, wave tile 32x32 (8 WMMA / 12 frag-loads per K-step).
// Kernel-B A-tile staged with GLOBAL_LOAD_ASYNC_TO_LDS when available.
// ---------------------------------------------------------------------------

typedef __attribute__((ext_vector_type(16))) __bf16        v16bf;
typedef __attribute__((ext_vector_type(4)))  __bf16        v4bf;
typedef __attribute__((ext_vector_type(8)))  float         v8f;
typedef __attribute__((ext_vector_type(4)))  float         v4f;
typedef __attribute__((ext_vector_type(4)))  unsigned int  v4u;
typedef __attribute__((ext_vector_type(4)))  int           v4i;

constexpr int kT    = 2048;   // tokens
constexpr int kH    = 1024;   // hidden
constexpr int kFFN  = 2048;   // ffn dim
constexpr int kTopK = 4;      // routed experts per token
constexpr int LSTR  = 40;     // LDS row stride (ushorts): 32 data + 8 pad (80B)

#if defined(__HIP_DEVICE_COMPILE__) && __has_builtin(__builtin_amdgcn_global_load_async_to_lds_b128)
#define USE_ASYNC_LDS 1
#else
#define USE_ASYNC_LDS 0
#endif

__device__ __forceinline__ unsigned short f2bf(float f) {
  unsigned int u = __float_as_uint(f);
  u = u + 0x7FFFu + ((u >> 16) & 1u);   // RNE
  return (unsigned short)(u >> 16);
}

// load float4 from global, convert via native bf16 cvt, store 8B to LDS
__device__ __forceinline__ void stage_f32_as_bf16(unsigned short* lds, const float* g) {
  v4f v = *(const v4f*)g;
  v4bf b;
  b.x = (__bf16)v.x; b.y = (__bf16)v.y; b.z = (__bf16)v.z; b.w = (__bf16)v.w;
  *(v4bf*)lds = b;
}

union FragAB { v4u u[2]; v16bf v; };

// A-matrix 16x32 bf16 fragment (ISA 7.12.2): lane L holds row M=L%16;
// VGPRs 0-3 -> K = kb..kb+7, VGPRs 4-7 -> K = 16+kb..16+kb+7, kb = (L<16?0:8).
__device__ __forceinline__ v16bf load_a_frag(const unsigned short* s, int mt, int lane) {
  const int row = mt * 16 + (lane & 15);
  const int kb  = (lane & 16) ? 8 : 0;
  const unsigned short* p = s + row * LSTR;
  FragAB f;
  f.u[0] = *(const v4u*)(p + kb);
  f.u[1] = *(const v4u*)(p + 16 + kb);
  return f.v;
}

// B-matrix 32x16 bf16 fragment: lane L holds col N=L%16;
// VGPRs 0-3 -> K = kb..kb+7, VGPRs 4-7 -> K = 16+kb..16+kb+7, kb = (L<16?0:8) (row-contiguous [n][k]).
__device__ __forceinline__ v16bf load_b_frag(const unsigned short* s, int nt, int lane) {
  const int row = nt * 16 + (lane & 15);
  const int ko  = (lane & 16) ? 8 : 0;
  const unsigned short* p = s + row * LSTR + ko;
  FragAB f;
  f.u[0] = *(const v4u*)(p);
  f.u[1] = *(const v4u*)(p + 16);
  return f.v;
}

#define WMMA_BF16(A, B, C) \
  __builtin_amdgcn_wmma_f32_16x16x32_bf16(false, (A), false, (B), (short)0, (C), false, false)

// ---------------------------------------------------------------------------
// Kernel 1: Hact = silu(X Wg^T) * (X Wu^T)   [bf16 out, one expert]
// Block = 128 tokens x 64 ffn-cols, 8 wave32 in a 4(M) x 2(N) grid,
// wave tile 32x32 -> 4 gate + 4 up accumulators, 8 WMMA per K-step.
// ---------------------------------------------------------------------------
__global__ __launch_bounds__(256) void moe_gateup(const float* __restrict__ X,
                                                  const float* __restrict__ Wgu,
                                                  unsigned short* __restrict__ Hact,
                                                  int e) {
  __shared__ unsigned short Xs[128 * LSTR];
  __shared__ unsigned short Gs[64 * LSTR];
  __shared__ unsigned short Us[64 * LSTR];

  const int tid  = threadIdx.x;
  const int lane = tid & 31;
  const int wave = tid >> 5;
  const int wm   = wave >> 1;   // 0..3 : 32-row slab
  const int wn   = wave & 1;    // 0..1 : 32-col slab
  const int trow = blockIdx.y * 128;
  const int ncol = blockIdx.x * 64;

  const float* Xp = X + (size_t)trow * kH;
  const float* Gp = Wgu + (size_t)e * 2 * kFFN * kH + (size_t)ncol * kH;
  const float* Up = Gp + (size_t)kFFN * kH;

  v8f cg[2][2] = {};
  v8f cu[2][2] = {};

  for (int k0 = 0; k0 < kH; k0 += 32) {
    // Stage X 128x32 (4 float4/thread) and Wg/Wu 64x32 (2 float4/thread each)
#pragma unroll
    for (int j = 0; j < 4; ++j) {
      const int q = tid + j * 256;
      const int r = q >> 3;
      const int c = (q & 7) * 4;
      stage_f32_as_bf16(&Xs[r * LSTR + c], Xp + (size_t)r * kH + k0 + c);
    }
#pragma unroll
    for (int j = 0; j < 2; ++j) {
      const int q = tid + j * 256;
      const int r = q >> 3;
      const int c = (q & 7) * 4;
      stage_f32_as_bf16(&Gs[r * LSTR + c], Gp + (size_t)r * kH + k0 + c);
      stage_f32_as_bf16(&Us[r * LSTR + c], Up + (size_t)r * kH + k0 + c);
      if (k0 + 32 < kH) {   // prefetch next K-slab of the streamed weights
        __builtin_prefetch(Gp + (size_t)r * kH + k0 + 32 + c, 0, 1);
        __builtin_prefetch(Up + (size_t)r * kH + k0 + 32 + c, 0, 1);
      }
    }
    __syncthreads();

    const v16bf a0 = load_a_frag(Xs, 2 * wm + 0, lane);
    const v16bf a1 = load_a_frag(Xs, 2 * wm + 1, lane);
    const v16bf g0 = load_b_frag(Gs, 2 * wn + 0, lane);
    const v16bf g1 = load_b_frag(Gs, 2 * wn + 1, lane);
    const v16bf u0 = load_b_frag(Us, 2 * wn + 0, lane);
    const v16bf u1 = load_b_frag(Us, 2 * wn + 1, lane);

    cg[0][0] = WMMA_BF16(a0, g0, cg[0][0]);
    cg[0][1] = WMMA_BF16(a0, g1, cg[0][1]);
    cg[1][0] = WMMA_BF16(a1, g0, cg[1][0]);
    cg[1][1] = WMMA_BF16(a1, g1, cg[1][1]);
    cu[0][0] = WMMA_BF16(a0, u0, cu[0][0]);
    cu[0][1] = WMMA_BF16(a0, u1, cu[0][1]);
    cu[1][0] = WMMA_BF16(a1, u0, cu[1][0]);
    cu[1][1] = WMMA_BF16(a1, u1, cu[1][1]);
    __syncthreads();
  }

  // Epilogue: h = silu(gate) * up, store bf16  (fast sigmoid via v_rcp_f32)
  unsigned short* Op = Hact + (size_t)trow * kFFN + ncol;
#pragma unroll
  for (int mi = 0; mi < 2; ++mi) {
    const int rb = wm * 32 + mi * 16 + ((lane & 16) ? 8 : 0);
#pragma unroll
    for (int ni = 0; ni < 2; ++ni) {
      const int cc = wn * 32 + ni * 16 + (lane & 15);
#pragma unroll
      for (int r = 0; r < 8; ++r) {
        const float g = cg[mi][ni][r];
        const float h = g * __builtin_amdgcn_rcpf(1.0f + __expf(-g)) * cu[mi][ni][r];
        Op[(size_t)(rb + r) * kFFN + cc] = f2bf(h);
      }
    }
  }
}

// ---------------------------------------------------------------------------
// Kernel 2: out (+)= w_e[t] * (Hact Wd^T)   [one expert; accum=0 writes]
// Same tiling; A-tile is already bf16 -> async DMA to LDS when available.
// ---------------------------------------------------------------------------
__global__ __launch_bounds__(256) void moe_down(const unsigned short* __restrict__ Hact,
                                                const float* __restrict__ Wd,
                                                const int* __restrict__ sel,
                                                const float* __restrict__ rw,
                                                float* __restrict__ out,
                                                int e, int eid, int accum) {
  __shared__ unsigned short As[128 * LSTR];
  __shared__ unsigned short Bs[64 * LSTR];
  __shared__ float wl[128];

  const int tid  = threadIdx.x;
  const int lane = tid & 31;
  const int wave = tid >> 5;
  const int wm   = wave >> 1;
  const int wn   = wave & 1;
  const int trow = blockIdx.y * 128;
  const int hcol = blockIdx.x * 64;

  // Collapsed routing weight w_e[t] for the 128 tokens of this tile
  if (tid < 128) {
    const int t = trow + tid;
    float w = 0.0f;
#pragma unroll
    for (int k = 0; k < kTopK; ++k) {
      w += (sel[t * kTopK + k] == eid) ? rw[t * kTopK + k] : 0.0f;
    }
    wl[tid] = w;
  }
  __syncthreads();

  const unsigned short* Ap = Hact + (size_t)trow * kFFN;
  const float*          Bp = Wd + (size_t)e * kH * kFFN + (size_t)hcol * kFFN;

  v8f c[2][2] = {};

  for (int k0 = 0; k0 < kFFN; k0 += 32) {
    // A tile: 128x32 bf16 = 512 x 16B chunks, 2 per thread (pure byte copy)
#pragma unroll
    for (int j = 0; j < 2; ++j) {
      const int q  = tid + j * 256;
      const int r  = q >> 2;
      const int cc = (q & 3) * 8;
#if USE_ASYNC_LDS
      __builtin_amdgcn_global_load_async_to_lds_b128(
          (v4i*)(Ap + (size_t)r * kFFN + k0 + cc),
          (v4i*)&As[r * LSTR + cc], 0, 0);
#else
      *(v4u*)&As[r * LSTR + cc] = *(const v4u*)(Ap + (size_t)r * kFFN + k0 + cc);
#endif
    }
    // B tile: Wd f32 -> bf16 (2 float4/thread)
#pragma unroll
    for (int j = 0; j < 2; ++j) {
      const int q = tid + j * 256;
      const int r = q >> 3;
      const int cc = (q & 7) * 4;
      stage_f32_as_bf16(&Bs[r * LSTR + cc], Bp + (size_t)r * kFFN + k0 + cc);
      if (k0 + 32 < kFFN) {
        __builtin_prefetch(Bp + (size_t)r * kFFN + k0 + 32 + cc, 0, 1);
      }
    }
#if USE_ASYNC_LDS
    asm volatile("s_wait_asynccnt 0x0" ::: "memory");
#endif
    __syncthreads();

    const v16bf a0 = load_a_frag(As, 2 * wm + 0, lane);
    const v16bf a1 = load_a_frag(As, 2 * wm + 1, lane);
    const v16bf b0 = load_b_frag(Bs, 2 * wn + 0, lane);
    const v16bf b1 = load_b_frag(Bs, 2 * wn + 1, lane);

    c[0][0] = WMMA_BF16(a0, b0, c[0][0]);
    c[0][1] = WMMA_BF16(a0, b1, c[0][1]);
    c[1][0] = WMMA_BF16(a1, b0, c[1][0]);
    c[1][1] = WMMA_BF16(a1, b1, c[1][1]);
    __syncthreads();
  }

  float* Op = out + (size_t)trow * kH + hcol;
#pragma unroll
  for (int mi = 0; mi < 2; ++mi) {
    const int rb = wm * 32 + mi * 16 + ((lane & 16) ? 8 : 0);
#pragma unroll
    for (int ni = 0; ni < 2; ++ni) {
      const int cc = wn * 32 + ni * 16 + (lane & 15);
#pragma unroll
      for (int r = 0; r < 8; ++r) {
        const float w = wl[rb + r];
        float v = w * c[mi][ni][r];
        const size_t idx = (size_t)(rb + r) * kH + cc;
        if (accum) v += Op[idx];
        Op[idx] = v;
      }
    }
  }
}

// ---------------------------------------------------------------------------
extern "C" void kernel_launch(void* const* d_in, const int* in_sizes, int n_in,
                              void* d_out, int out_size, void* d_ws, size_t ws_size,
                              hipStream_t stream) {
  (void)in_sizes; (void)n_in; (void)out_size; (void)ws_size;

  const float* hidden = (const float*)d_in[0];      // [T, H] f32
  const int*   sel    = (const int*)d_in[1];        // [T, K] i32
  const float* rw     = (const float*)d_in[2];      // [T, K] f32
  const float* gup    = (const float*)d_in[3];      // [4, 2*FFN, H] f32
  const float* down   = (const float*)d_in[4];      // [4, H, FFN] f32
  float* out = (float*)d_out;                       // [T, H] f32

  // Workspace: one expert's activation, bf16 [T, FFN] = 8 MiB (reused per expert)
  unsigned short* hact = (unsigned short*)d_ws;

  const dim3 blk(256);
  const dim3 gA(kFFN / 64, kT / 128);
  const dim3 gB(kH / 64, kT / 128);
  const int eids[4] = {0, 8, 16, 24};  // LOCAL_EXPERT_IDS (round-robin, ep_rank 0)

  for (int e = 0; e < 4; ++e) {
    moe_gateup<<<gA, blk, 0, stream>>>(hidden, gup, hact, e);
    moe_down<<<gB, blk, 0, stream>>>(hact, down, sel, rw, out, e, eids[e], e > 0);
  }
}
